// EqualizedOddsLoss_59072980189339
// MI455X (gfx1250) — 8-bit matrix operand order repaired and checked by reading, compiled, not run on hardware
//
#include <hip/hip_runtime.h>
#include <hip/hip_bf16.h>

typedef __attribute__((ext_vector_type(8))) int   v8i;
typedef __attribute__((ext_vector_type(4))) float f32x4;
typedef __attribute__((ext_vector_type(4))) int   i32x4;

#define NGRP 8
#define NCOL 5

__global__ void eo_init_kernel(int* __restrict__ ws) {
    int i = threadIdx.x;
    if (i < NGRP * NCOL + 1) ws[i] = 0;
}

__global__ __launch_bounds__(256)
void eo_hist_kernel(const float* __restrict__ pred,
                    const float* __restrict__ lab,
                    const int*   __restrict__ grp,
                    int* __restrict__ counts,
                    int* __restrict__ condflag,
                    int nIter) {
    __shared__ unsigned int smem[8 * 64];
    __shared__ int blkAux[NGRP * NCOL + 1];

    const int tid  = threadIdx.x;
    const int lane = tid & 31;
    const int warp = tid >> 5;
    const int hi   = lane >> 4;
    const int m    = lane & 15;
    const unsigned amask = (m < NGRP) ? 0x01010101u : 0u;
    const unsigned bmask = (m < NCOL) ? 0x01010101u : 0u;

    if (tid <= NGRP * NCOL) blkAux[tid] = 0;
    __syncthreads();

    unsigned int* gdwA = &smem[warp * 64];
    unsigned int* fdwA = gdwA + 32;

    const int wave   = (int)(blockIdx.x * blockDim.x + tid) >> 5;
    const int nWaves = (int)(gridDim.x * blockDim.x) >> 5;

    v8i acc = {};
    bool cond = false;

    for (int it = wave; it < nIter; it += nWaves) {
        const long e = (long)it * 128 + lane * 4;
        const f32x4 p4 = __builtin_nontemporal_load(reinterpret_cast<const f32x4*>(pred + e));
        const f32x4 l4 = __builtin_nontemporal_load(reinterpret_cast<const f32x4*>(lab  + e));
        const i32x4 g4 = __builtin_nontemporal_load(reinterpret_cast<const i32x4*>(grp  + e));

        unsigned fpk = 0, gpk = 0;
        #pragma unroll
        for (int j = 0; j < 4; ++j) {
            const float p = p4[j];
            const bool  L  = l4[j] > 0.5f;
            const bool  b0 = p > 0.0f;
            const bool  b5 = p > 0.5f;
            const unsigned fb = L ? ((b0 ? 1u : 0u) | (b5 ? 4u : 0u) | 16u)
                                  : ((b0 ? 2u : 0u) | (b5 ? 8u : 0u));
            fpk |= fb << (8 * j);
            gpk |= 1u << (((unsigned)g4[j] & 7u) + 8u * j);
            cond |= (p < 0.0f) | (p > 1.0f);
        }
        gdwA[lane] = gpk;
        fdwA[lane] = fpk;
        asm volatile("" ::: "memory");

        #pragma unroll
        for (int s = 0; s < 2; ++s) {
            const unsigned* gdw = gdwA + s * 16;
            const unsigned* fdw = fdwA + s * 16;
            v8i A, Bm;
            #pragma unroll
            for (int v = 0; v < 8; ++v) {
                const int kA = ((v >> 1) << 4) | ((v & 1) << 2) | (hi << 3);
                A[v] = (int)((gdw[kA >> 2] >> m) & amask);
                const int kB = ((v >> 1) << 4) | ((v & 1) << 2) | (hi << 3);
                Bm[v] = (int)((fdw[kB >> 2] >> m) & bmask);
            }
            acc = __builtin_amdgcn_wmma_i32_16x16x64_iu8(
                      false, A, false, Bm, acc, false, false);
        }
        asm volatile("" ::: "memory");
    }

    if (hi == 0 && m < NCOL) {
        #pragma unroll
        for (int r = 0; r < NGRP; ++r)
            atomicAdd(&blkAux[r * NCOL + m], acc[r]);
    }
    if (cond) blkAux[NGRP * NCOL] = 1;
    __syncthreads();

    if (tid < NGRP * NCOL) {
        const int v = blkAux[tid];
        if (v) atomicAdd(&counts[tid], v);
    } else if (tid == NGRP * NCOL) {
        if (blkAux[NGRP * NCOL]) atomicOr(condflag, 1);
    }
}

__global__ void eo_tail_kernel(const float* __restrict__ pred,
                               const float* __restrict__ lab,
                               const int*   __restrict__ grp,
                               int* __restrict__ counts,
                               int* __restrict__ condflag,
                               int start, int n) {
    int i = start + (int)(blockIdx.x * blockDim.x + threadIdx.x);
    if (i >= n) return;
    const float p = pred[i];
    const bool  L  = lab[i] > 0.5f;
    const int   g  = grp[i] & 7;
    const bool  b0 = p > 0.0f, b5 = p > 0.5f;
    if ( L && b0) atomicAdd(&counts[g * NCOL + 0], 1);
    if (!L && b0) atomicAdd(&counts[g * NCOL + 1], 1);
    if ( L && b5) atomicAdd(&counts[g * NCOL + 2], 1);
    if (!L && b5) atomicAdd(&counts[g * NCOL + 3], 1);
    if ( L)       atomicAdd(&counts[g * NCOL + 4], 1);
    if (p < 0.0f || p > 1.0f) atomicOr(condflag, 1);
}

__global__ void eo_final_kernel(const int* __restrict__ counts,
                                const int* __restrict__ condflag,
                                float* __restrict__ out, float ntot) {
    if (threadIdx.x != 0 || blockIdx.x != 0) return;
    const bool sig = (*condflag) != 0;
    float tpr[NGRP], fpr[NGRP];
    #pragma unroll
    for (int g = 0; g < NGRP; ++g) {
        const float tp  = (float)(sig ? counts[g * NCOL + 0] : counts[g * NCOL + 2]);
        const float fp  = (float)(sig ? counts[g * NCOL + 1] : counts[g * NCOL + 3]);
        const float pos = (float)counts[g * NCOL + 4];
        const float neg = ntot - pos;
        tpr[g] = tp / (pos + 1e-8f);
        fpr[g] = fp / (neg + 1e-8f);
    }
    float tot = 0.0f;
    #pragma unroll
    for (int i = 0; i < NGRP; ++i)
        #pragma unroll
        for (int j = i + 1; j < NGRP; ++j)
            tot += fabsf(tpr[i] - tpr[j]) + fabsf(fpr[i] - fpr[j]);
    out[0] = 1.0f * tot;
}

extern "C" void kernel_launch(void* const* d_in, const int* in_sizes, int n_in,
                              void* d_out, int out_size, void* d_ws, size_t ws_size,
                              hipStream_t stream) {
    const float* pred = (const float*)d_in[0];
    const float* lab  = (const float*)d_in[1];
    const int*   grp  = (const int*)d_in[2];
    const int n = in_sizes[0];

    int* counts = (int*)d_ws;
    int* flag   = counts + NGRP * NCOL;

    eo_init_kernel<<<1, 64, 0, stream>>>(counts);

    const int nIter = n / 128;
    if (nIter > 0) {
        eo_hist_kernel<<<1024, 256, 0, stream>>>(pred, lab, grp, counts, flag, nIter);
    }
    const int rem = n - nIter * 128;
    if (rem > 0) {
        eo_tail_kernel<<<1, 128, 0, stream>>>(pred, lab, grp, counts, flag,
                                              nIter * 128, n);
    }
    eo_final_kernel<<<1, 32, 0, stream>>>(counts, flag, (float*)d_out, (float)n);
}
